// RWKVv7Separator_68762426409178
// MI455X (gfx1250) — matrix operand order repaired and checked by reading, not compiled, not run on hardware
//
#include <hip/hip_runtime.h>
#include <hip/hip_bf16.h>
#include <math.h>

typedef __attribute__((ext_vector_type(16))) __bf16 v16bf;
typedef __attribute__((ext_vector_type(8)))  __bf16 v8bf;
typedef __attribute__((ext_vector_type(4)))  __bf16 v4bf;
typedef __attribute__((ext_vector_type(8)))  float  v8f;

#define BM 64
#define BN 64
#define BK 32
#define LDT 40   // padded LDS row stride: 80B rows keep 16B fragment alignment
                 // (80*r % 16 == 0) and spread rows across banks (20 dwords)

// ---------------------------------------------------------------------------
// Generic GEMM: out[M,N] = ACT( (A (+ Ashift*mix)) @ W + bias )
//  A: [M,K] f32, W: [K,N] f32 (row-major); staged to LDS as bf16 with
//  double-buffering; accumulated in f32 via v_wmma_f32_16x16x32_bf16.
//  ACT: 0=none, 1=tanh, 2=sigmoid, 3=relu^2.  All flags compile-time.
// ---------------------------------------------------------------------------
template<bool HAS_MIX, int ACT, bool NGUARD>
__global__ __launch_bounds__(128) void gemm_bf16_wmma(
    const float* __restrict__ A, const float* __restrict__ Ashift,
    const float* __restrict__ mix, const float* __restrict__ W,
    const float* __restrict__ bias, float* __restrict__ out,
    int M, int N, int K)
{
  __shared__ __bf16 Al[2][BM * LDT];
  __shared__ __bf16 Bl[2][BN * LDT];
  const int tid  = threadIdx.x;
  const int lane = tid & 31;
  const int wave = tid >> 5;          // 4 waves, each owns a 16-row strip
  const int m0 = blockIdx.x * BM;
  const int n0 = blockIdx.y * BN;

  v8f acc[4] = {};

  const int row  = lane & 15;
  const int half = lane >> 4;

  // register staging for the in-flight tile (distinct regs -> loads overlap)
  float4 aR[4], sR[4], mR[4];
  float  wR[4][4];

  auto load_tile = [&](int kb) {
#pragma unroll
    for (int it = 0; it < 4; ++it) {
      int idx = tid + it * 128;
      int r  = idx >> 3;               // tile row (8 quads per row)
      int c4 = (idx & 7) * 4;          // col quad
      const size_t ao = (size_t)(m0 + r) * K + kb + c4;
      aR[it] = *(const float4*)&A[ao];
      if (HAS_MIX) {
        sR[it] = *(const float4*)&Ashift[ao];
        mR[it] = *(const float4*)&mix[kb + c4];
      }
    }
#pragma unroll
    for (int it = 0; it < 4; ++it) {
      int idx = tid + it * 128;
      int n  = idx >> 3;               // 0..63
      int k4 = (idx & 7) * 4;          // 0,4,...,28
      int gn = n0 + n;
      if (!NGUARD || gn < N) {
        const float* wp = &W[(size_t)(kb + k4) * N + gn];
        wR[it][0] = wp[0];
        wR[it][1] = wp[(size_t)N];
        wR[it][2] = wp[(size_t)2 * N];
        wR[it][3] = wp[(size_t)3 * N];
      } else {
        wR[it][0] = 0.f; wR[it][1] = 0.f; wR[it][2] = 0.f; wR[it][3] = 0.f;
      }
    }
  };

  auto store_tile = [&](int buf) {
#pragma unroll
    for (int it = 0; it < 4; ++it) {
      int idx = tid + it * 128;
      int r  = idx >> 3;
      int c4 = (idx & 7) * 4;
      float vx = aR[it].x, vy = aR[it].y, vz = aR[it].z, vw = aR[it].w;
      if (HAS_MIX) {
        vx += sR[it].x * mR[it].x; vy += sR[it].y * mR[it].y;
        vz += sR[it].z * mR[it].z; vw += sR[it].w * mR[it].w;
      }
      v4bf o;
      o[0] = (__bf16)vx; o[1] = (__bf16)vy; o[2] = (__bf16)vz; o[3] = (__bf16)vw;
      *(v4bf*)&Al[buf][r * LDT + c4] = o;
    }
#pragma unroll
    for (int it = 0; it < 4; ++it) {
      int idx = tid + it * 128;
      int n  = idx >> 3;
      int k4 = (idx & 7) * 4;
      v4bf o;
      o[0] = (__bf16)wR[it][0]; o[1] = (__bf16)wR[it][1];
      o[2] = (__bf16)wR[it][2]; o[3] = (__bf16)wR[it][3];
      *(v4bf*)&Bl[buf][n * LDT + k4] = o;
    }
  };

  // prologue: tile 0 into buffer 0
  load_tile(0);
  store_tile(0);

  for (int kb = 0; kb < K; kb += BK) {
    const int buf = (kb / BK) & 1;
    __syncthreads();                   // buf stores visible; buf^1 reads done
    const bool more = (kb + BK) < K;
    if (more) load_tile(kb + BK);      // global loads in flight under the math

    // A fragment 16x32 bf16 (ISA 7.12.2): lane<16 holds K[0:8]+K[16:24] of its
    // row, lane>=16 holds K[8:16]+K[24:32]
    const __bf16* arow = &Al[buf][(wave * 16 + row) * LDT];
    v8bf alo = *(const v8bf*)&arow[half * 8];
    v8bf ahi = *(const v8bf*)&arow[16 + half * 8];
    v16bf af;
#pragma unroll
    for (int i = 0; i < 8; ++i) { af[i] = alo[i]; af[i + 8] = ahi[i]; }

    // B fragment 32x16: lane<16 holds K[0:8]+K[16:24] of its column, lane>=16 K[8:16]+K[24:32]
#pragma unroll
    for (int j = 0; j < 4; ++j) {
      const __bf16* brow = &Bl[buf][(j * 16 + row) * LDT];
      v8bf blo = *(const v8bf*)&brow[half * 8];
      v8bf bhi = *(const v8bf*)&brow[16 + half * 8];
      v16bf bf;
#pragma unroll
      for (int i = 0; i < 8; ++i) { bf[i] = blo[i]; bf[i + 8] = bhi[i]; }
      acc[j] = __builtin_amdgcn_wmma_f32_16x16x32_bf16(
          false, af, false, bf, (short)0, acc[j], false, false);
    }

    if (more) store_tile(buf ^ 1);     // cvt + ds_store after wmma issue
  }

  // D layout: VGPR r -> (M=r, N=lane) lanes 0-15; (M=8+r, N=lane-16) lanes 16-31
#pragma unroll
  for (int j = 0; j < 4; ++j) {
    int n = n0 + j * 16 + row;
    if (NGUARD && n >= N) continue;
    float bv = bias ? bias[n] : 0.0f;
#pragma unroll
    for (int r = 0; r < 8; ++r) {
      int m = m0 + wave * 16 + half * 8 + r;
      float v = acc[j][r] + bv;
      if (ACT == 1)      v = tanhf(v);
      else if (ACT == 2) v = 1.0f / (1.0f + expf(-v));
      else if (ACT == 3) { v = fmaxf(v, 0.0f); v = v * v; }
      out[(size_t)m * N + n] = v;
    }
  }
}

// ---------------------------------------------------------------------------
// LayerNorm over last dim (one block per row)
// ---------------------------------------------------------------------------
__global__ __launch_bounds__(256) void ln_kernel(
    const float* __restrict__ x, const float* __restrict__ w,
    const float* __restrict__ b, float* __restrict__ out, int C, float eps)
{
  __shared__ float red[256];
  const int row = blockIdx.x;
  const float* xr = x + (size_t)row * C;
  float* orow = out + (size_t)row * C;
  float s = 0.f, s2 = 0.f;
  for (int c = threadIdx.x; c < C; c += 256) { float v = xr[c]; s += v; s2 += v * v; }
  red[threadIdx.x] = s; __syncthreads();
  for (int off = 128; off >= 1; off >>= 1) {
    if (threadIdx.x < off) red[threadIdx.x] += red[threadIdx.x + off];
    __syncthreads();
  }
  float mean = red[0] / (float)C; __syncthreads();
  red[threadIdx.x] = s2; __syncthreads();
  for (int off = 128; off >= 1; off >>= 1) {
    if (threadIdx.x < off) red[threadIdx.x] += red[threadIdx.x + off];
    __syncthreads();
  }
  float var = red[0] / (float)C - mean * mean;
  float rstd = rsqrtf(var + eps);
  for (int c = threadIdx.x; c < C; c += 256)
    orow[c] = (xr[c] - mean) * rstd * w[c] + b[c];
}

// per-64-channel (head) group sum; 512-thread blocks
__device__ __forceinline__ float head_sum(float* red, int c, float v)
{
  red[c] = v;
  __syncthreads();
  for (int off = 32; off >= 1; off >>= 1) {
    if ((c & 63) < off) red[c] += red[c + off];
    __syncthreads();
  }
  float s = red[c & ~63];
  __syncthreads();
  return s;
}

// ---------------------------------------------------------------------------
// Pre-WKV fusion: head-normalized kk, decay, a-gate, v_first mixing, k scaling
// ---------------------------------------------------------------------------
__global__ __launch_bounds__(512) void prep_kernel(
    float* __restrict__ kbuf, float* __restrict__ vbuf,
    const float* __restrict__ wlo, const float* __restrict__ alo,
    const float* __restrict__ vlo, float* __restrict__ vfirst,
    const float* __restrict__ k_k, const float* __restrict__ k_a,
    const float* __restrict__ w0, const float* __restrict__ a0,
    const float* __restrict__ v0,
    float* __restrict__ wd, float* __restrict__ aab, float* __restrict__ bbb,
    int layer)
{
  __shared__ float red[512];
  const int row = blockIdx.x, c = threadIdx.x;
  const size_t idx = (size_t)row * 512 + c;
  float k = kbuf[idx];
  float kkv = k * k_k[c];
  float ss = head_sum(red, c, kkv * kkv);
  kkv = kkv / fmaxf(sqrtf(ss), 1e-12f);
  float a = 1.0f / (1.0f + expf(-(a0[c] + alo[idx])));
  float wv = w0[c] + wlo[idx];
  float wlog = -log1pf(expf(-wv)) - 0.5f;      // -softplus(-wv) - 0.5
  float decay = expf(-expf(wlog));
  float v = vbuf[idx];
  if (layer == 0) {
    vfirst[idx] = v;
  } else {
    float gate = 1.0f / (1.0f + expf(-(v0[c] + vlo[idx])));
    v = v + (vfirst[idx] - v) * gate;
  }
  float k2 = k * (1.0f + (a - 1.0f) * k_a[c]);
  wd[idx]  = decay;
  aab[idx] = -kkv;
  bbb[idx] = kkv * a;
  kbuf[idx] = k2;
  vbuf[idx] = v;
}

// ---------------------------------------------------------------------------
// WKV7 scan: one block per (batch, head); 64x64 f32 state lives in VGPRs
// (16 elems/thread). thread -> (row i = tid/4, cols j0..j0+15 = (tid&3)*16).
// Row reductions across 4 lanes via wave32 shfl_xor.
// ---------------------------------------------------------------------------
__global__ __launch_bounds__(256) void wkv_kernel(
    const float* __restrict__ r, const float* __restrict__ wdec,
    const float* __restrict__ k, const float* __restrict__ v,
    const float* __restrict__ aa, const float* __restrict__ bb,
    float* __restrict__ y, int Tlen)
{
  const int bh = blockIdx.x;           // 0..15
  const int b = bh >> 3, h = bh & 7;
  const int tid = threadIdx.x;
  const int i  = tid >> 2;             // state row 0..63
  const int j0 = (tid & 3) * 16;       // state col block
  float S[16];
#pragma unroll
  for (int j = 0; j < 16; ++j) S[j] = 0.f;

  __shared__ float sr[64], sw[64], sk[64], sv[64], sa[64], sb[64];

  for (int t = 0; t < Tlen; ++t) {
    const size_t base = ((size_t)(b * Tlen + t)) * 512 + h * 64;
    if (tid < 64) {
      sr[tid] = r[base + tid];  sw[tid] = wdec[base + tid];
      sk[tid] = k[base + tid];  sv[tid] = v[base + tid];
      sa[tid] = aa[base + tid]; sb[tid] = bb[base + tid];
    }
    __syncthreads();

    float sap = 0.f;
#pragma unroll
    for (int j = 0; j < 16; ++j) sap += S[j] * sa[j0 + j];
    sap += __shfl_xor(sap, 1, 32);
    sap += __shfl_xor(sap, 2, 32);

    const float vi = sv[i];
    float yp = 0.f;
#pragma unroll
    for (int j = 0; j < 16; ++j) {
      float s = S[j] * sw[j0 + j] + sap * sb[j0 + j] + vi * sk[j0 + j];
      S[j] = s;
      yp += s * sr[j0 + j];
    }
    yp += __shfl_xor(yp, 1, 32);
    yp += __shfl_xor(yp, 2, 32);
    if ((tid & 3) == 0) y[base + i] = yp;
    __syncthreads();
  }
}

// ---------------------------------------------------------------------------
// Post-WKV: per-head GroupNorm + (r.k.r_k).sum * v residual + g gate
// ---------------------------------------------------------------------------
__global__ __launch_bounds__(512) void post_kernel(
    const float* __restrict__ y, const float* __restrict__ r,
    const float* __restrict__ k2, const float* __restrict__ v2,
    const float* __restrict__ g, const float* __restrict__ gnw,
    const float* __restrict__ gnb, const float* __restrict__ r_k,
    float* __restrict__ out)
{
  __shared__ float red[512];
  const int row = blockIdx.x, c = threadIdx.x;
  const int h = c >> 6, n = c & 63;
  const size_t idx = (size_t)row * 512 + c;
  float yv = y[idx];
  float mean = head_sum(red, c, yv) * (1.0f / 64.0f);
  float var  = head_sum(red, c, yv * yv) * (1.0f / 64.0f) - mean * mean;
  float gn = (yv - mean) * rsqrtf(var + 64e-5f) * gnw[c] + gnb[c];
  float dot = head_sum(red, c, r[idx] * k2[idx] * r_k[h * 64 + n]);
  out[idx] = (gn + dot * v2[idx]) * g[idx];
}

// ---------------------------------------------------------------------------
// Elementwise helpers
// ---------------------------------------------------------------------------
__global__ void shiftdiff_kernel(const float* __restrict__ src,
                                 float* __restrict__ dst,
                                 int Tlen, int C, int total)
{
  int idx = blockIdx.x * blockDim.x + threadIdx.x;
  if (idx >= total) return;
  int t = (idx / C) % Tlen;
  float prev = (t > 0) ? src[idx - C] : 0.0f;
  dst[idx] = prev - src[idx];
}

__global__ void add_kernel(float* __restrict__ x, const float* __restrict__ o,
                           int total)
{
  int idx = blockIdx.x * blockDim.x + threadIdx.x;
  if (idx >= total) return;
  x[idx] += o[idx];
}

__global__ void flip_kernel(const float* __restrict__ src,
                            float* __restrict__ dst, int Tlen, int C, int total)
{
  int idx = blockIdx.x * blockDim.x + threadIdx.x;
  if (idx >= total) return;
  int c = idx % C; int bt = idx / C; int t = bt % Tlen; int b = bt / Tlen;
  dst[idx] = src[((size_t)(b * Tlen + (Tlen - 1 - t))) * C + c];
}

__global__ void combine_kernel(const float* __restrict__ xf,
                               const float* __restrict__ xb,
                               float* __restrict__ hout,
                               int Tlen, int C, int total)
{
  int idx = blockIdx.x * blockDim.x + threadIdx.x;
  if (idx >= total) return;
  int c = idx % C; int bt = idx / C; int t = bt % Tlen; int b = bt / Tlen;
  float bv = xb[((size_t)(b * Tlen + (Tlen - 1 - t))) * C + c];
  hout[idx] = 0.5f * (xf[idx] + bv);
}

__global__ void softmax2_kernel(const float* __restrict__ hm,
                                float* __restrict__ out, int total, int Cin)
{
  int idx = blockIdx.x * blockDim.x + threadIdx.x;
  if (idx >= total) return;
  int m = idx / Cin, c = idx % Cin;
  float l0 = hm[(size_t)m * (2 * Cin) + 2 * c];
  float l1 = hm[(size_t)m * (2 * Cin) + 2 * c + 1];
  float p0 = 1.0f / (1.0f + expf(l1 - l0));
  out[idx] = p0;
  out[(size_t)total + idx] = 1.0f - p0;
}

// ---------------------------------------------------------------------------
// Host orchestration
// ---------------------------------------------------------------------------
extern "C" void kernel_launch(void* const* d_in, const int* in_sizes, int n_in,
                              void* d_out, int out_size, void* d_ws, size_t ws_size,
                              hipStream_t stream)
{
  (void)in_sizes; (void)n_in; (void)out_size; (void)ws_size;
  const int Bc = 2, Tc = 512, CIN = 1024, HID = 512, NLAYER = 6;
  const int M = Bc * Tc;                 // 1024 rows
  const int TOT = M * HID;               // 524288
  const int TOTC = M * CIN;              // 1048576

  // ---- inputs (setup_inputs insertion order, recursively) ----
  const float* z_mix   = (const float*)d_in[0];
  const float* down_w  = (const float*)d_in[1];
  const float* down_b  = (const float*)d_in[2];
  const float* up_w    = (const float*)d_in[3];
  const float* up_b    = (const float*)d_in[4];
  const float* hm_ln_w = (const float*)d_in[5];
  const float* hm_ln_b = (const float*)d_in[6];
  const float* hm_w    = (const float*)d_in[7];
  const float* hm_b    = (const float*)d_in[8];
  const int FWD_BASE = 9, BWD_BASE = 42;  // 33 stacked leaves each

  enum {L_LN1W=0,L_LN1B,L_LN2W,L_LN2B,L_XR,L_XW,L_XK,L_XV,L_XA,L_XG,
        L_W0,L_W1,L_W2,L_A0,L_A1,L_A2,L_V0,L_V1,L_V2,L_G1,L_G2,
        L_KK,L_KA,L_RK,L_WR,L_WK,L_WV,L_WO,L_GNW,L_GNB,L_CXK,L_CWK,L_CWV};
  static const size_t leafSize[33] = {
    512,512,512,512, 512,512,512,512,512,512,
    512, 16384,16384, 512,16384,16384, 512,16384,16384,
    49152,49152, 512,512, 512,
    262144,262144,262144,262144, 512,512, 512, 1048576,1048576};

  // ---- workspace carve-up (floats) ----
  float* ws = (float*)d_ws;
  size_t off = 0;
  auto alloc = [&](size_t n) { float* p = ws + off; off += n; return p; };
  float* X     = alloc(TOT);   // current activations (fwd dir / fwd result)
  float* XSAVE = alloc(TOT);   // post-down activations (pristine)
  float* XB    = alloc(TOT);   // bwd-direction activations
  float* LNb   = alloc(TOT);
  float* XX    = alloc(TOT);
  float* Rb    = alloc(TOT);
  float* Kb    = alloc(TOT);
  float* Vb    = alloc(TOT);
  float* WD    = alloc(TOT);
  float* AA    = alloc(TOT);
  float* BBu   = alloc(TOT);
  float* Gb    = alloc(TOT);
  float* Yb    = alloc(TOT);
  float* Pb    = alloc(TOT);
  float* Ob    = alloc(TOT);
  float* VF    = alloc(TOT);
  float* WBUF  = alloc(TOT);
  float* ABUF  = alloc(TOT);
  float* VBUF  = alloc(TOT);
  float* HB    = alloc(TOT);
  float* LB1   = alloc((size_t)M * 96);
  float* H1    = alloc((size_t)M * 2048);
  float* UP    = alloc((size_t)M * CIN);
  float* HL    = alloc((size_t)M * CIN);
  float* HMb   = alloc((size_t)M * 2048);

  // compile-time-dispatched GEMM launcher
  auto gemm = [&](const float* A, const float* Ash, const float* mixv,
                  const float* W, const float* bias, float* outp,
                  int N, int K, int act) {
    dim3 grid(M / BM, (N + BN - 1) / BN);
    const bool guard = (N % BN) != 0;
    if (!mixv) {
      gemm_bf16_wmma<false,0,false><<<grid,128,0,stream>>>(A,Ash,mixv,W,bias,outp,M,N,K);
    } else if (!guard) {
      if (act == 3)
        gemm_bf16_wmma<true,3,false><<<grid,128,0,stream>>>(A,Ash,mixv,W,bias,outp,M,N,K);
      else
        gemm_bf16_wmma<true,0,false><<<grid,128,0,stream>>>(A,Ash,mixv,W,bias,outp,M,N,K);
    } else {
      if (act == 1)
        gemm_bf16_wmma<true,1,true><<<grid,128,0,stream>>>(A,Ash,mixv,W,bias,outp,M,N,K);
      else if (act == 2)
        gemm_bf16_wmma<true,2,true><<<grid,128,0,stream>>>(A,Ash,mixv,W,bias,outp,M,N,K);
      else
        gemm_bf16_wmma<true,0,true><<<grid,128,0,stream>>>(A,Ash,mixv,W,bias,outp,M,N,K);
    }
  };
  auto egrid = [](int n) { return (n + 255) / 256; };

  auto run_core = [&](int SB, float* Xp) {
    auto P = [&](int leaf, int layer) -> const float* {
      return (const float*)d_in[SB + leaf] + (size_t)layer * leafSize[leaf];
    };
    for (int L = 0; L < NLAYER; ++L) {
      // ---- tmix ----
      ln_kernel<<<M, 256, 0, stream>>>(Xp, P(L_LN1W,L), P(L_LN1B,L), LNb, HID, 1e-5f);
      shiftdiff_kernel<<<egrid(TOT), 256, 0, stream>>>(LNb, XX, Tc, HID, TOT);
      gemm(LNb, XX, P(L_XR,L), P(L_WR,L), nullptr, Rb, 512, 512, 0);
      gemm(LNb, XX, P(L_XK,L), P(L_WK,L), nullptr, Kb, 512, 512, 0);
      gemm(LNb, XX, P(L_XV,L), P(L_WV,L), nullptr, Vb, 512, 512, 0);
      gemm(LNb, XX, P(L_XW,L), P(L_W1,L), nullptr, LB1, 32, 512, 1);   // tanh
      gemm(LB1, nullptr, nullptr, P(L_W2,L), nullptr, WBUF, 512, 32, 0);
      gemm(LNb, XX, P(L_XA,L), P(L_A1,L), nullptr, LB1, 32, 512, 0);
      gemm(LB1, nullptr, nullptr, P(L_A2,L), nullptr, ABUF, 512, 32, 0);
      if (L > 0) {
        gemm(LNb, XX, P(L_XV,L), P(L_V1,L), nullptr, LB1, 32, 512, 0);
        gemm(LB1, nullptr, nullptr, P(L_V2,L), nullptr, VBUF, 512, 32, 0);
      }
      gemm(LNb, XX, P(L_XG,L), P(L_G1,L), nullptr, LB1, 96, 512, 2);   // sigmoid
      gemm(LB1, nullptr, nullptr, P(L_G2,L), nullptr, Gb, 512, 96, 0);
      prep_kernel<<<M, 512, 0, stream>>>(Kb, Vb, WBUF, ABUF, VBUF, VF,
          P(L_KK,L), P(L_KA,L), P(L_W0,L), P(L_A0,L), P(L_V0,L),
          WD, AA, BBu, L);
      wkv_kernel<<<16, 256, 0, stream>>>(Rb, WD, Kb, Vb, AA, BBu, Yb, Tc);
      post_kernel<<<M, 512, 0, stream>>>(Yb, Rb, Kb, Vb, Gb,
          P(L_GNW,L), P(L_GNB,L), P(L_RK,L), Pb);
      gemm(Pb, nullptr, nullptr, P(L_WO,L), nullptr, Ob, 512, 512, 0);
      add_kernel<<<egrid(TOT), 256, 0, stream>>>(Xp, Ob, TOT);
      // ---- cmix ----
      ln_kernel<<<M, 256, 0, stream>>>(Xp, P(L_LN2W,L), P(L_LN2B,L), LNb, HID, 1e-5f);
      shiftdiff_kernel<<<egrid(TOT), 256, 0, stream>>>(LNb, XX, Tc, HID, TOT);
      gemm(LNb, XX, P(L_CXK,L), P(L_CWK,L), nullptr, H1, 2048, 512, 3); // relu^2
      gemm(H1, nullptr, nullptr, P(L_CWV,L), nullptr, Ob, 512, 2048, 0);
      add_kernel<<<egrid(TOT), 256, 0, stream>>>(Xp, Ob, TOT);
    }
  };

  // down-projection
  gemm(z_mix, nullptr, nullptr, down_w, down_b, X, HID, CIN, 0);
  hipMemcpyAsync(XSAVE, X, (size_t)TOT * sizeof(float),
                 hipMemcpyDeviceToDevice, stream);
  // forward core (mutates X in place)
  run_core(FWD_BASE, X);
  // backward core on time-flipped input
  flip_kernel<<<egrid(TOT), 256, 0, stream>>>(XSAVE, XB, Tc, HID, TOT);
  run_core(BWD_BASE, XB);
  // h = 0.5 * (xf + flip(xb))
  combine_kernel<<<egrid(TOT), 256, 0, stream>>>(X, XB, HB, Tc, HID, TOT);
  // up-projection + head
  gemm(HB, nullptr, nullptr, up_w, up_b, UP, CIN, HID, 0);
  ln_kernel<<<M, 256, 0, stream>>>(UP, hm_ln_w, hm_ln_b, HL, CIN, 1e-5f);
  gemm(HL, nullptr, nullptr, hm_w, hm_b, HMb, 2 * CIN, CIN, 0);
  softmax2_kernel<<<egrid(TOTC), 256, 0, stream>>>(HMb, (float*)d_out, TOTC, CIN);
}
